// HSELoss_65154653880591
// MI455X (gfx1250) — matrix operand order repaired and checked by reading, compiled, not run on hardware
//
#include <hip/hip_runtime.h>
#include <hip/hip_bf16.h>
#include <math.h>

typedef __attribute__((ext_vector_type(16))) _Float16 v16h;
typedef __attribute__((ext_vector_type(8)))  _Float16 v8h;
typedef __attribute__((ext_vector_type(4)))  _Float16 v4h;
typedef __attribute__((ext_vector_type(8)))  float    v8f;
typedef __attribute__((ext_vector_type(4)))  float    v4f;

#define N_ROWS     8192
#define DIMS       128
#define IPC_N      512
#define QBASE      (N_ROWS - IPC_N)   // 7680
#define EASY_TILES 64                 // cols [0,1024)  -> tiles [0,64)
#define HARD_END   480                // cols [1024,7680) -> tiles [64,480); [480,512) excluded
#define SIGMA_INV  10.0f

// Monotone order-preserving encoding for values in [-2, 2]: v+4 in [2,6] (positive
// floats compare like their bit patterns), so atomicMax on uint == float max.
static __device__ __forceinline__ unsigned enc_ord(float v) { return __float_as_uint(v + 4.0f); }
static __device__ __forceinline__ float    dec_ord(unsigned u) { return __uint_as_float(u) - 4.0f; }

// ---------------------------------------------------------------------------
// Kernel A: L2-normalize each row of fvec, store as f16; init the max arrays.
// One wave per row (wave32: 4 floats/lane), 8 rows per 256-thread block.
// ---------------------------------------------------------------------------
__global__ void __launch_bounds__(256)
normalize_kernel(const float* __restrict__ fvec, _Float16* __restrict__ fh,
                 unsigned* __restrict__ gE, unsigned* __restrict__ gH)
{
    const int wave = threadIdx.x >> 5;
    const int lane = threadIdx.x & 31;
    const int row  = blockIdx.x * 8 + wave;

    // Init per-query running-max slots (first two blocks cover 512 entries).
    const int gid = blockIdx.x * 256 + threadIdx.x;
    if (gid < IPC_N) {
        const unsigned e = enc_ord(-2.0f);
        gE[gid] = e;
        gH[gid] = e;
    }

    const float* src = fvec + (size_t)row * DIMS + lane * 4;
    v4f v = *(const v4f*)src;
    float s = v.x * v.x + v.y * v.y + v.z * v.z + v.w * v.w;
#pragma unroll
    for (int m = 16; m >= 1; m >>= 1) s += __shfl_xor(s, m, 32);

    const float inv = 1.0f / fmaxf(sqrtf(s), 1e-12f);
    v4h o;
    o.x = (_Float16)(v.x * inv);
    o.y = (_Float16)(v.y * inv);
    o.z = (_Float16)(v.z * inv);
    o.w = (_Float16)(v.w * inv);
    *(v4h*)(fh + (size_t)row * DIMS + lane * 4) = o;
}

// ---------------------------------------------------------------------------
// Kernel B: WMMA similarity GEMM + segmented row-max.
// grid = (32 m-tiles, 2 n-halves), block = 256 threads = 8 waves.
// Each wave processes an ADJACENT PAIR of 16x16 output tiles per iteration:
// two independent WMMA accumulator chains (8 interleaved WMMAs/iter) so no
// WMMA depends on its immediate predecessor, and 16 b128 loads are in flight
// per wait. Pair stride-16 partition: wave w handles nt = ybase + 2w + 16i,
// i = 0..14 (30 tiles). The easy/hard boundary (tile 64) is even, so a pair
// never straddles it -> loop is split into a pure-easy and a pure-hard phase
// (no per-iteration region branch).
// ---------------------------------------------------------------------------

// One tile-pair: 8 B-fragment loads, 8 interleaved WMMAs, fold into RUN[8].
#define DO_PAIR(ITER, RUN)                                                              \
    {                                                                                   \
        const int nt = ybase + 2 * wave + 16 * (ITER);                                  \
        const _Float16* pb = F + (size_t)(nt * 16 + r) * DIMS + koff;                   \
        __builtin_prefetch(pb + 16 * 16 * DIMS, 0, 1); /* next iteration's pair */      \
        v16h b00 = *(const v16h*)(pb);                                                  \
        v16h b01 = *(const v16h*)(pb + 32);                                             \
        v16h b02 = *(const v16h*)(pb + 64);                                             \
        v16h b03 = *(const v16h*)(pb + 96);                                             \
        const _Float16* pc = pb + 16 * DIMS;                                            \
        v16h b10 = *(const v16h*)(pc);                                                  \
        v16h b11 = *(const v16h*)(pc + 32);                                             \
        v16h b12 = *(const v16h*)(pc + 64);                                             \
        v16h b13 = *(const v16h*)(pc + 96);                                             \
        v8f c0 = {}, c1 = {};                                                           \
        c0 = __builtin_amdgcn_wmma_f32_16x16x32_f16(false, afr[0], false, b00, (short)0, c0, false, false); \
        c1 = __builtin_amdgcn_wmma_f32_16x16x32_f16(false, afr[0], false, b10, (short)0, c1, false, false); \
        c0 = __builtin_amdgcn_wmma_f32_16x16x32_f16(false, afr[1], false, b01, (short)0, c0, false, false); \
        c1 = __builtin_amdgcn_wmma_f32_16x16x32_f16(false, afr[1], false, b11, (short)0, c1, false, false); \
        c0 = __builtin_amdgcn_wmma_f32_16x16x32_f16(false, afr[2], false, b02, (short)0, c0, false, false); \
        c1 = __builtin_amdgcn_wmma_f32_16x16x32_f16(false, afr[2], false, b12, (short)0, c1, false, false); \
        c0 = __builtin_amdgcn_wmma_f32_16x16x32_f16(false, afr[3], false, b03, (short)0, c0, false, false); \
        c1 = __builtin_amdgcn_wmma_f32_16x16x32_f16(false, afr[3], false, b13, (short)0, c1, false, false); \
        _Pragma("unroll")                                                               \
        for (int j = 0; j < 8; ++j) RUN[j] = fmaxf(RUN[j], fmaxf(c0[j], c1[j]));        \
    }

__global__ void __launch_bounds__(256)
sim_kernel(const _Float16* __restrict__ F,
           unsigned* __restrict__ gE, unsigned* __restrict__ gH)
{
    const int wave  = threadIdx.x >> 5;
    const int lane  = threadIdx.x & 31;
    const int r     = lane & 15;    // row-in-tile (A) / col-in-tile (B)
    const int h     = lane >> 4;    // half-wave select
    const int mtile = blockIdx.x;   // 0..31 -> queries QBASE + mtile*16 ..
    const int ybase = blockIdx.y * 240;

    // --- A fragments (reused across all 30 tiles): 16x32 f16, contiguous k order (matches B):
    // lane l (l<16): row M=l, K=0..15 in v0..7; lane l+16: K=16..31.
    const _Float16* qrow = F + (size_t)(QBASE + mtile * 16 + r) * DIMS;
    v16h afr[4];
#pragma unroll
    for (int kb = 0; kb < 4; ++kb) {
        const _Float16* pa = qrow + kb * 32 + h * 16;
        v8h lo = *(const v8h*)(pa);
        v8h hi = *(const v8h*)(pa + 8);
#pragma unroll
        for (int i = 0; i < 8; ++i) { afr[kb][i] = lo[i]; afr[kb][i + 8] = hi[i]; }
    }

    float runE[8], runH[8];
#pragma unroll
    for (int j = 0; j < 8; ++j) { runE[j] = -2.0f; runH[j] = -2.0f; }

    const int koff = h * 16;   // B layout: lanes 16-31 hold K=16..31

    // Phase 1 (easy region, only in the first n-half): i = 0..3 -> tiles 0..63.
    const int iEasyEnd = (blockIdx.y == 0) ? 4 : 0;
    for (int i = 0; i < iEasyEnd; ++i) DO_PAIR(i, runE)
    // Phase 2 (hard region): remaining iterations -> tiles [64,480).
    for (int i = iEasyEnd; i < 15; ++i) DO_PAIR(i, runH)

    // Butterfly max across the 16-lane half groups (col dimension of the tile).
#pragma unroll
    for (int m = 1; m <= 8; m <<= 1) {
#pragma unroll
        for (int j = 0; j < 8; ++j) {
            runE[j] = fmaxf(runE[j], __shfl_xor(runE[j], m, 32));
            runH[j] = fmaxf(runH[j], __shfl_xor(runH[j], m, 32));
        }
    }

    // C layout: VGPR j, lanes 0-15 -> row j; lanes 16-31 -> row j+8.
    if (r == 0) {
        const int q0 = mtile * 16 + h * 8;
#pragma unroll
        for (int j = 0; j < 8; ++j) {
            atomicMax(&gE[q0 + j], enc_ord(runE[j]));
            atomicMax(&gH[q0 + j], enc_ord(runH[j]));
        }
    }
}

// ---------------------------------------------------------------------------
// Kernel C: per-query loss + deterministic tree reduction, single block.
// loss_q = log(1 + exp((VCH-VCE)/sigma)); out = a * mean(loss).
// ---------------------------------------------------------------------------
__global__ void __launch_bounds__(256)
loss_kernel(const unsigned* __restrict__ gE, const unsigned* __restrict__ gH,
            const float* __restrict__ a, float* __restrict__ out)
{
    __shared__ float sred[256];
    const int t = threadIdx.x;
    float sum = 0.0f;
    for (int q = t; q < IPC_N; q += 256) {
        const float e = dec_ord(gE[q]);
        const float hm = dec_ord(gH[q]);
        const float z = (hm - e) * SIGMA_INV;
        sum += fmaxf(z, 0.0f) + log1pf(expf(-fabsf(z)));   // stable log(1+e^z)
    }
    sred[t] = sum;
    __syncthreads();
#pragma unroll
    for (int s = 128; s > 0; s >>= 1) {
        if (t < s) sred[t] += sred[t + s];
        __syncthreads();
    }
    if (t == 0) out[0] = a[0] * (sred[0] / (float)IPC_N);
}

// ---------------------------------------------------------------------------
extern "C" void kernel_launch(void* const* d_in, const int* in_sizes, int n_in,
                              void* d_out, int out_size, void* d_ws, size_t ws_size,
                              hipStream_t stream)
{
    const float* fvec = (const float*)d_in[0];
    // d_in[1] = Lvec (unused by the reference loss value)
    const float* a = (const float*)d_in[2];
    // d_in[3] = IPC (512), d_in[4] = bz (8192): fixed by the problem, hardcoded.

    _Float16* fh = (_Float16*)d_ws;                                  // 8192*128 f16 = 2 MB
    unsigned* gE = (unsigned*)((char*)d_ws + (size_t)N_ROWS * DIMS * sizeof(_Float16));
    unsigned* gH = gE + IPC_N;
    float* out = (float*)d_out;

    normalize_kernel<<<N_ROWS / 8, 256, 0, stream>>>(fvec, fh, gE, gH);
    dim3 g(32, 2);
    sim_kernel<<<g, 256, 0, stream>>>(fh, gE, gH);
    loss_kernel<<<1, 256, 0, stream>>>(gE, gH, a, out);
}
